// ParallelMLPAttention_64536178590037
// MI455X (gfx1250) — compile-verified, matrix operand order checked by reading, not run on hardware
//
#include <hip/hip_runtime.h>

// ---------------------------------------------------------------------------
// Types for CDNA5 WMMA (wave32): v_wmma_f32_16x16x32_bf16
// ---------------------------------------------------------------------------
typedef __attribute__((ext_vector_type(16))) __bf16 v16bf;
typedef __attribute__((ext_vector_type(8)))  __bf16 v8bf;
typedef __attribute__((ext_vector_type(8)))  float  v8f;

#define HIDDEN 1024
#define NHEAD  16
#define HDIM   64
#define MLPH   4096
#define SEQ    2048
#define BATCH  2
#define MROWS  (BATCH * SEQ)          // 4096 tokens
#define N1     (3 * HIDDEN + MLPH)    // 7168
#define CONCATW (HIDDEN + MLPH)       // 5120

__device__ __forceinline__ v16bf pack16(v8bf lo, v8bf hi) {
    v16bf r;
#pragma unroll
    for (int i = 0; i < 8; ++i) { r[i] = lo[i]; r[i + 8] = hi[i]; }
    return r;
}

// Loads one 16x32 bf16 A/B fragment from LDS / global.
// p already includes the half-wave K offset (half*8); second chunk is +16 elems.
__device__ __forceinline__ v16bf ldfrag(const __bf16* p) {
    return pack16(*(const v8bf*)p, *(const v8bf*)(p + 16));
}

__device__ __forceinline__ unsigned pack_bf16x2(float a, float b) {
    union { __bf16 h[2]; unsigned u; } x;
    x.h[0] = (__bf16)a; x.h[1] = (__bf16)b;
    return x.u;
}

// Fast tanh-GELU: tanh(u) = 1 - 2/(exp(2u)+1), exp via v_exp_f32.
__device__ __forceinline__ float gelu_tanh(float v) {
    const float c = 0.7978845608028654f;
    float u = c * (v + 0.044715f * v * v * v);
    float t = 1.0f - 2.0f / (__expf(2.0f * u) + 1.0f);
    return 0.5f * v * (1.0f + t);
}

// ---------------------------------------------------------------------------
// Shared 128x128 macro-tile GEMM mainloop, K step 32, 8 waves (2x4),
// each wave owns a 64x32 region = 4x2 accumulator tiles.
// A: MxK row-major bf16. Bt: NxK row-major bf16 (i.e., B transposed).
// Software-pipelined: next tile's global loads issue while WMMAs run.
// ---------------------------------------------------------------------------
template <int KD>
__device__ __forceinline__ void gemm_loop(const __bf16* __restrict__ A,
                                          const __bf16* __restrict__ Bt,
                                          int m0, int n0,
                                          __bf16* shA, __bf16* shB,
                                          v8f (&acc)[4][2]) {
    const int tid  = threadIdx.x;
    const int lane = tid & 31;
    const int half = lane >> 4;
    const int r    = lane & 15;
    const int wave = tid >> 5;
    const int wm   = (wave >> 2) * 64;
    const int wn   = (wave & 3) * 32;

    // Each thread owns 2 16B chunks of the A tile and 2 of the B tile.
    const int row0 = tid >> 2,          cc0 = tid & 3;
    const int row1 = (tid + 256) >> 2,  cc1 = (tid + 256) & 3;

    const __bf16* gA0 = &A[(size_t)(m0 + row0) * KD + cc0 * 8];
    const __bf16* gA1 = &A[(size_t)(m0 + row1) * KD + cc1 * 8];
    const __bf16* gB0 = &Bt[(size_t)(n0 + row0) * KD + cc0 * 8];
    const __bf16* gB1 = &Bt[(size_t)(n0 + row1) * KD + cc1 * 8];

    uint4 ra0 = *(const uint4*)gA0, ra1 = *(const uint4*)gA1;
    uint4 rb0 = *(const uint4*)gB0, rb1 = *(const uint4*)gB1;

    for (int k0 = 0; k0 < KD; k0 += 32) {
        __syncthreads();
        *(uint4*)(&shA[row0 * 40 + cc0 * 8]) = ra0;
        *(uint4*)(&shA[row1 * 40 + cc1 * 8]) = ra1;
        *(uint4*)(&shB[row0 * 40 + cc0 * 8]) = rb0;
        *(uint4*)(&shB[row1 * 40 + cc1 * 8]) = rb1;
        __syncthreads();

        if (k0 + 32 < KD) {   // prefetch next K tile; waits deferred
            ra0 = *(const uint4*)(gA0 + k0 + 32);
            ra1 = *(const uint4*)(gA1 + k0 + 32);
            rb0 = *(const uint4*)(gB0 + k0 + 32);
            rb1 = *(const uint4*)(gB1 + k0 + 32);
        }

        v16bf af[4], bfr[2];
#pragma unroll
        for (int mt = 0; mt < 4; ++mt)
            af[mt] = ldfrag(&shA[(wm + 16 * mt + r) * 40 + half * 8]);
#pragma unroll
        for (int nt = 0; nt < 2; ++nt)
            bfr[nt] = ldfrag(&shB[(wn + 16 * nt + r) * 40 + half * 8]);
#pragma unroll
        for (int mt = 0; mt < 4; ++mt)
#pragma unroll
            for (int nt = 0; nt < 2; ++nt)
                acc[mt][nt] = __builtin_amdgcn_wmma_f32_16x16x32_bf16(
                    false, af[mt], false, bfr[nt], (short)0, acc[mt][nt],
                    false, false);
    }
}

// ---------------------------------------------------------------------------
// GEMM1: h = x @ W1 + b1.  Cols < 3072 -> f32 qkv scratch.
//        Cols >= 3072 -> gelu -> bf16 into concat buffer (offset 1024).
// ---------------------------------------------------------------------------
__global__ __launch_bounds__(256) void gemm1_kernel(
    const __bf16* __restrict__ A, const __bf16* __restrict__ Bt,
    const float* __restrict__ bias, float* __restrict__ qkv,
    __bf16* __restrict__ concat) {
    __shared__ __bf16 shA[128 * 40];
    __shared__ __bf16 shB[128 * 40];
    const int m0 = blockIdx.y * 128, n0 = blockIdx.x * 128;
    v8f acc[4][2];
#pragma unroll
    for (int mt = 0; mt < 4; ++mt)
#pragma unroll
        for (int nt = 0; nt < 2; ++nt) acc[mt][nt] = v8f{};

    gemm_loop<HIDDEN>(A, Bt, m0, n0, shA, shB, acc);

    const int lane = threadIdx.x & 31, half = lane >> 4, r = lane & 15;
    const int wave = threadIdx.x >> 5;
    const int wm = (wave >> 2) * 64, wn = (wave & 3) * 32;
#pragma unroll
    for (int mt = 0; mt < 4; ++mt)
#pragma unroll
        for (int nt = 0; nt < 2; ++nt)
#pragma unroll
            for (int i = 0; i < 8; ++i) {
                int row = m0 + wm + 16 * mt + i + half * 8;
                int col = n0 + wn + 16 * nt + r;
                float v = acc[mt][nt][i] + bias[col];
                if (col < 3 * HIDDEN) {
                    qkv[(size_t)row * (3 * HIDDEN) + col] = v;
                } else {
                    concat[(size_t)row * CONCATW + (col - 2 * HIDDEN)] =
                        (__bf16)gelu_tanh(v);
                }
            }
}

// ---------------------------------------------------------------------------
// GEMM2: out = concat @ W2 + b2 (f32 output)
// ---------------------------------------------------------------------------
__global__ __launch_bounds__(256) void gemm2_kernel(
    const __bf16* __restrict__ A, const __bf16* __restrict__ Bt,
    const float* __restrict__ bias, float* __restrict__ out) {
    __shared__ __bf16 shA[128 * 40];
    __shared__ __bf16 shB[128 * 40];
    const int m0 = blockIdx.y * 128, n0 = blockIdx.x * 128;
    v8f acc[4][2];
#pragma unroll
    for (int mt = 0; mt < 4; ++mt)
#pragma unroll
        for (int nt = 0; nt < 2; ++nt) acc[mt][nt] = v8f{};

    gemm_loop<CONCATW>(A, Bt, m0, n0, shA, shB, acc);

    const int lane = threadIdx.x & 31, half = lane >> 4, r = lane & 15;
    const int wave = threadIdx.x >> 5;
    const int wm = (wave >> 2) * 64, wn = (wave & 3) * 32;
#pragma unroll
    for (int mt = 0; mt < 4; ++mt)
#pragma unroll
        for (int nt = 0; nt < 2; ++nt)
#pragma unroll
            for (int i = 0; i < 8; ++i) {
                int row = m0 + wm + 16 * mt + i + half * 8;
                int col = n0 + wn + 16 * nt + r;
                out[(size_t)row * HIDDEN + col] = acc[mt][nt][i] + bias[col];
            }
}

// ---------------------------------------------------------------------------
// RMSNorm + RoPE + bf16 pack into [B,H,L,64] Q/K/V buffers.
// One wave per (token, head); lane = rotation pair p (d = 2p, 2p+1).
// ---------------------------------------------------------------------------
__global__ __launch_bounds__(256) void qkvprep_kernel(
    const float* __restrict__ qkv, const float* __restrict__ pe,
    const float* __restrict__ qs, const float* __restrict__ ks,
    __bf16* __restrict__ Q, __bf16* __restrict__ Kb,
    __bf16* __restrict__ Vb) {
    const int tid = threadIdx.x;
    const int lane = tid & 31, wave = tid >> 5;
    const int rg = blockIdx.x;               // token 0..4095
    const int h = blockIdx.y * 8 + wave;     // head 0..15
    const int b = rg >> 11, l = rg & (SEQ - 1);
    const int p = lane, d0 = 2 * p, d1 = 2 * p + 1;

    const size_t base = (size_t)rg * (3 * HIDDEN) + h * HDIM;
    float2 qv = *(const float2*)(&qkv[base + d0]);
    float2 kv = *(const float2*)(&qkv[base + HIDDEN + d0]);
    float2 vv = *(const float2*)(&qkv[base + 2 * HIDDEN + d0]);
    float q0 = qv.x, q1 = qv.y, k0 = kv.x, k1 = kv.y;

    float sq = q0 * q0 + q1 * q1;
    float sk = k0 * k0 + k1 * k1;
#pragma unroll
    for (int m = 1; m < 32; m <<= 1) {
        sq += __shfl_xor(sq, m, 32);
        sk += __shfl_xor(sk, m, 32);
    }
    const float rq = rsqrtf(sq * (1.0f / HDIM) + 1e-6f);
    const float rk = rsqrtf(sk * (1.0f / HDIM) + 1e-6f);
    q0 *= rq * qs[d0]; q1 *= rq * qs[d1];
    k0 *= rk * ks[d0]; k1 *= rk * ks[d1];

    const float4 e = *(const float4*)(pe + ((size_t)l * 32 + p) * 4);  // [l][p][2][2]

    const size_t ob = ((size_t)(b * NHEAD + h) * SEQ + l) * HDIM;
    *(unsigned*)(&Q[ob + d0])  = pack_bf16x2(e.x * q0 + e.y * q1,
                                             e.z * q0 + e.w * q1);
    *(unsigned*)(&Kb[ob + d0]) = pack_bf16x2(e.x * k0 + e.y * k1,
                                             e.z * k0 + e.w * k1);
    *(unsigned*)(&Vb[ob + d0]) = pack_bf16x2(vv.x, vv.y);
}

// ---------------------------------------------------------------------------
// Flash attention: block = 4 waves, each wave owns 16 q rows x full d=64.
// Loops over 64-key tiles with online softmax; WMMA for QK^T and PV.
// K/V tiles are register-prefetched one iteration ahead.
// Writes bf16 attention output into concat[:, 0:1024].
// ---------------------------------------------------------------------------
__global__ __launch_bounds__(128) void attn_kernel(
    const __bf16* __restrict__ Q, const __bf16* __restrict__ Kk,
    const __bf16* __restrict__ V, __bf16* __restrict__ concat) {
    __shared__ __bf16 shK[64 * 72];       // [key][d]
    __shared__ __bf16 shVt[64 * 72];      // [d][key]
    __shared__ __bf16 shP[4 * 16 * 72];   // per-wave P strip [row][key]

    const int tid = threadIdx.x;
    const int lane = tid & 31, wave = tid >> 5;
    const int half = lane >> 4, r = lane & 15;
    const int bh = blockIdx.y;            // b*16 + h
    const int b = bh >> 4, h = bh & 15;
    const size_t hb = (size_t)bh * SEQ * HDIM;
    const int qrow0 = blockIdx.x * 64 + wave * 16;

    // Q fragments: 16 rows x 64 d = 2 K-step fragments, loaded once.
    v16bf qf[2];
#pragma unroll
    for (int kk = 0; kk < 2; ++kk) {
        const __bf16* p = Q + hb + (size_t)(qrow0 + r) * HDIM + kk * 32 + half * 8;
        qf[kk] = ldfrag(p);
    }

    // Each thread owns 4 16B chunks of the 64x64 K tile (and V tile).
    int crow[4], ccc[4];
#pragma unroll
    for (int j = 0; j < 4; ++j) {
        int cid = tid + 128 * j;          // 0..511
        crow[j] = cid >> 3; ccc[j] = cid & 7;
    }

    uint4 kreg[4], vreg[4];
#pragma unroll
    for (int j = 0; j < 4; ++j) {
        kreg[j] = *(const uint4*)(Kk + hb + (size_t)crow[j] * HDIM + ccc[j] * 8);
        vreg[j] = *(const uint4*)(V  + hb + (size_t)crow[j] * HDIM + ccc[j] * 8);
    }

    v8f o[4];
#pragma unroll
    for (int nt = 0; nt < 4; ++nt) o[nt] = v8f{};
    float mrow[8], lrow[8];
#pragma unroll
    for (int i = 0; i < 8; ++i) { mrow[i] = -1e30f; lrow[i] = 0.0f; }

    for (int kt = 0; kt < SEQ / 64; ++kt) {
        __syncthreads();
        // Commit prefetched K tile (row-major) and V tile (transposed).
#pragma unroll
        for (int j = 0; j < 4; ++j)
            *(uint4*)(&shK[crow[j] * 72 + ccc[j] * 8]) = kreg[j];
#pragma unroll
        for (int j = 0; j < 4; ++j) {
            union { uint4 u; v8bf bv; } cv; cv.u = vreg[j];
#pragma unroll
            for (int e = 0; e < 8; ++e)
                shVt[(ccc[j] * 8 + e) * 72 + crow[j]] = cv.bv[e];
        }
        __syncthreads();

        if (kt + 1 < SEQ / 64) {          // prefetch next tile; waits deferred
            const size_t nb = hb + (size_t)(kt + 1) * 64 * HDIM;
#pragma unroll
            for (int j = 0; j < 4; ++j) {
                kreg[j] = *(const uint4*)(Kk + nb + (size_t)crow[j] * HDIM + ccc[j] * 8);
                vreg[j] = *(const uint4*)(V  + nb + (size_t)crow[j] * HDIM + ccc[j] * 8);
            }
        }

        // scores = Q K^T  (4 n-tiles x 2 k-steps)
        v8f s[4];
#pragma unroll
        for (int nt = 0; nt < 4; ++nt) s[nt] = v8f{};
#pragma unroll
        for (int nt = 0; nt < 4; ++nt)
#pragma unroll
            for (int kk = 0; kk < 2; ++kk) {
                v16bf kf = ldfrag(&shK[(16 * nt + r) * 72 + kk * 32 + half * 8]);
                s[nt] = __builtin_amdgcn_wmma_f32_16x16x32_bf16(
                    false, qf[kk], false, kf, (short)0, s[nt], false, false);
            }

        // Online softmax. Row m = i + 8*half lives in one 16-lane half,
        // so row reductions are shuffles with masks 1,2,4,8.
#pragma unroll
        for (int i = 0; i < 8; ++i) {
            float rmax = -1e30f;
#pragma unroll
            for (int nt = 0; nt < 4; ++nt) {
                s[nt][i] *= 0.125f;  // HEAD_DIM^-0.5
                rmax = fmaxf(rmax, s[nt][i]);
            }
#pragma unroll
            for (int m = 1; m < 16; m <<= 1)
                rmax = fmaxf(rmax, __shfl_xor(rmax, m, 32));
            const float mnew = fmaxf(mrow[i], rmax);
            const float corr = __expf(mrow[i] - mnew);
            float rsum = 0.0f;
#pragma unroll
            for (int nt = 0; nt < 4; ++nt) {
                float pv = __expf(s[nt][i] - mnew);
                s[nt][i] = pv;
                rsum += pv;
            }
#pragma unroll
            for (int m = 1; m < 16; m <<= 1) rsum += __shfl_xor(rsum, m, 32);
            lrow[i] = lrow[i] * corr + rsum;
            mrow[i] = mnew;
#pragma unroll
            for (int nt = 0; nt < 4; ++nt) o[nt][i] *= corr;
        }

        // C-layout -> A-layout for P via per-wave LDS strip (DS in-order per
        // wave: stores are seen by the subsequent loads without a barrier).
        __bf16* pw = &shP[wave * 16 * 72];
#pragma unroll
        for (int nt = 0; nt < 4; ++nt)
#pragma unroll
            for (int i = 0; i < 8; ++i)
                pw[(i + half * 8) * 72 + 16 * nt + r] = (__bf16)s[nt][i];

        v16bf pf[2];
#pragma unroll
        for (int kk = 0; kk < 2; ++kk)
            pf[kk] = ldfrag(&pw[r * 72 + kk * 32 + half * 8]);

        // O += P V
#pragma unroll
        for (int nt = 0; nt < 4; ++nt)
#pragma unroll
            for (int kk = 0; kk < 2; ++kk) {
                v16bf vf = ldfrag(&shVt[(16 * nt + r) * 72 + kk * 32 + half * 8]);
                o[nt] = __builtin_amdgcn_wmma_f32_16x16x32_bf16(
                    false, pf[kk], false, vf, (short)0, o[nt], false, false);
            }
    }

    // Normalize and store bf16 into concat[:, h*64 + d].
    const int grow0 = b * SEQ + blockIdx.x * 64 + wave * 16;
#pragma unroll
    for (int nt = 0; nt < 4; ++nt)
#pragma unroll
        for (int i = 0; i < 8; ++i) {
            int row = grow0 + i + half * 8;
            int col = h * HDIM + 16 * nt + r;
            concat[(size_t)row * CONCATW + col] = (__bf16)(o[nt][i] / lrow[i]);
        }
}

// ---------------------------------------------------------------------------
// fp32 -> bf16 elementwise convert (4 elements / thread)
// ---------------------------------------------------------------------------
__global__ __launch_bounds__(256) void cvt_kernel(const float* __restrict__ s,
                                                  __bf16* __restrict__ d,
                                                  int n4) {
    int i = blockIdx.x * blockDim.x + threadIdx.x;
    if (i < n4) {
        float4 f = *(const float4*)(s + (size_t)i * 4);
        uint2 o;
        o.x = pack_bf16x2(f.x, f.y);
        o.y = pack_bf16x2(f.z, f.w);
        *(uint2*)(d + (size_t)i * 4) = o;
    }
}

// ---------------------------------------------------------------------------
// Tiled transpose + convert: src[R][C] f32 -> dst[C][R] bf16
// ---------------------------------------------------------------------------
__global__ __launch_bounds__(256) void transpose_cvt_kernel(
    const float* __restrict__ src, __bf16* __restrict__ dst, int R, int C) {
    __shared__ float tile[32][33];
    const int tx = threadIdx.x, ty = threadIdx.y;
    const int c0 = blockIdx.x * 32, r0 = blockIdx.y * 32;
#pragma unroll
    for (int j = 0; j < 32; j += 8)
        tile[ty + j][tx] = src[(size_t)(r0 + ty + j) * C + c0 + tx];
    __syncthreads();
#pragma unroll
    for (int j = 0; j < 32; j += 8)
        dst[(size_t)(c0 + ty + j) * R + r0 + tx] = (__bf16)tile[tx][ty + j];
}

// ---------------------------------------------------------------------------
// Launch
// ---------------------------------------------------------------------------
extern "C" void kernel_launch(void* const* d_in, const int* in_sizes, int n_in,
                              void* d_out, int out_size, void* d_ws,
                              size_t ws_size, hipStream_t stream) {
    (void)in_sizes; (void)n_in; (void)out_size; (void)ws_size;
    const float* x  = (const float*)d_in[0];
    const float* pe = (const float*)d_in[1];
    const float* W1 = (const float*)d_in[2];
    const float* b1 = (const float*)d_in[3];
    const float* W2 = (const float*)d_in[4];
    const float* b2 = (const float*)d_in[5];
    const float* qs = (const float*)d_in[6];
    const float* ks = (const float*)d_in[7];
    float* out = (float*)d_out;

    char* ws = (char*)d_ws;
    __bf16* x_bf   = (__bf16*)(ws);                 //  8 MiB
    __bf16* W1t    = (__bf16*)(ws + 8388608);       // 14 MiB (7168x1024)
    __bf16* W2t    = (__bf16*)(ws + 23068672);      // 10 MiB (1024x5120)
    float*  qkv    = (float*) (ws + 33554432);      // 48 MiB (4096x3072 f32)
    __bf16* concat = (__bf16*)(ws + 83886080);      // 40 MiB (4096x5120)
    __bf16* Qb     = (__bf16*)(ws + 125829120);     //  8 MiB
    __bf16* Kb     = (__bf16*)(ws + 134217728);     //  8 MiB
    __bf16* Vb     = (__bf16*)(ws + 142606336);     //  8 MiB

    cvt_kernel<<<(MROWS * HIDDEN / 4 + 255) / 256, 256, 0, stream>>>(
        x, x_bf, MROWS * HIDDEN / 4);
    transpose_cvt_kernel<<<dim3(N1 / 32, HIDDEN / 32), dim3(32, 8), 0, stream>>>(
        W1, W1t, HIDDEN, N1);
    transpose_cvt_kernel<<<dim3(HIDDEN / 32, CONCATW / 32), dim3(32, 8), 0,
                           stream>>>(W2, W2t, CONCATW, HIDDEN);

    gemm1_kernel<<<dim3(N1 / 128, MROWS / 128), 256, 0, stream>>>(
        x_bf, W1t, b1, qkv, concat);

    qkvprep_kernel<<<dim3(MROWS, 2), 256, 0, stream>>>(qkv, pe, qs, ks, Qb, Kb,
                                                       Vb);

    attn_kernel<<<dim3(SEQ / 64, BATCH * NHEAD), 128, 0, stream>>>(Qb, Kb, Vb,
                                                                   concat);

    gemm2_kernel<<<dim3(HIDDEN / 128, MROWS / 128), 256, 0, stream>>>(
        concat, W2t, b2, out);
}
